// SEKernelAttention_82343112999022
// MI455X (gfx1250) — compile-verified, matrix operand order checked by reading, not run on hardware
//
#include <hip/hip_runtime.h>
#include <cmath>
#include <stdint.h>

typedef _Float16 h4  __attribute__((ext_vector_type(4)));
typedef _Float16 h8  __attribute__((ext_vector_type(8)));
typedef _Float16 h16 __attribute__((ext_vector_type(16)));
typedef float    f8  __attribute__((ext_vector_type(8)));
typedef unsigned int u32x4 __attribute__((ext_vector_type(4)));
typedef int      i32x4 __attribute__((ext_vector_type(4)));
typedef int      i32x8 __attribute__((ext_vector_type(8)));

// LDS row pitches (halves); multiples of 8 halves = 16B so b128 DS ops stay aligned.
#define QP 72
#define KP 72
#define VP 40
#define PP 40

// Raw hardware transcendentals: args here are never denormal / don't need the
// IEEE fixup sequences the libm versions expand to (~10 VALU ops per call).
__device__ __forceinline__ float fast_sqrt(float x) { return __builtin_amdgcn_sqrtf(x); }
__device__ __forceinline__ float fast_exp2(float x) { return __builtin_amdgcn_exp2f(x); }
__device__ __forceinline__ float fast_rcp(float x)  { return __builtin_amdgcn_rcpf(x); }

__device__ __forceinline__ h16 ldfrag(const _Float16* base, int o0, int o1) {
  h8 a = *(const h8*)(base + o0);
  h8 b = *(const h8*)(base + o1);
  h16 r;
#pragma unroll
  for (int j = 0; j < 8; ++j) { r[j] = a[j]; r[j + 8] = b[j]; }
  return r;
}

__device__ __forceinline__ f8 zero8() {
  f8 z;
#pragma unroll
  for (int i = 0; i < 8; ++i) z[i] = 0.f;
  return z;
}

// Tensor Data Mover: 2D tile (f16, row-major, stride0 elements/row) global -> LDS,
// with hardware row padding reproducing our padded LDS pitch.
// pad_int_code: interval = 2^code * 8 bytes before each pad; pad_amt_code: (code+1) DWORDs pad.
__device__ __forceinline__ void tdm_load_2d(uint32_t lds_addr, const void* gptr,
                                            uint32_t tdim0, uint32_t tdim1,
                                            uint32_t tile0, uint32_t tile1,
                                            uint32_t stride0,
                                            uint32_t pad_int_code, uint32_t pad_amt_code)
{
  uint64_t ga = (uint64_t)(uintptr_t)gptr;
  u32x4 g0;
  g0[0] = 1u;                                                  // count=1 (valid user D#)
  g0[1] = lds_addr;                                            // LDS byte offset
  g0[2] = (uint32_t)ga;                                        // global_addr[31:0]
  g0[3] = (uint32_t)((ga >> 32) & 0x01FFFFFFu) | (2u << 30);   // global_addr[56:32] | type=2
  i32x8 g1;
  g1[0] = (int)((1u << 16)                       // data_size = 2 bytes
              | (1u << 20)                       // pad_enable
              | (pad_int_code << 22)             // pad_interval
              | (pad_amt_code << 25));           // pad_amount
  g1[1] = (int)((tdim0 & 0xFFFFu) << 16);                               // tensor_dim0[15:0]
  g1[2] = (int)(((tdim0 >> 16) & 0xFFFFu) | ((tdim1 & 0xFFFFu) << 16)); // dim0 hi | dim1 lo
  g1[3] = (int)(((tdim1 >> 16) & 0xFFFFu) | ((tile0 & 0xFFFFu) << 16)); // dim1 hi | tile_dim0
  g1[4] = (int)(tile1 & 0xFFFFu);                                       // tile_dim1 | tile_dim2=0
  g1[5] = (int)stride0;                                                 // tensor_dim0_stride lo
  g1[6] = 0;                                                            // stride0 hi | dim1_stride lo
  g1[7] = 0;
  i32x4 z4 = {0, 0, 0, 0};
#if __clang_major__ >= 23
  i32x8 z8 = {0, 0, 0, 0, 0, 0, 0, 0};
  __builtin_amdgcn_tensor_load_to_lds(g0, g1, z4, z4, z8, 0);
#else
  __builtin_amdgcn_tensor_load_to_lds(g0, g1, z4, z4, 0);
#endif
}

// One-shot f32 -> f16 pre-conversion of Q, K, V into workspace (TDM path).
__global__ __launch_bounds__(256) void cvt3_kernel(
    const float* __restrict__ Q, const float* __restrict__ K, const float* __restrict__ V,
    _Float16* __restrict__ Qh, _Float16* __restrict__ Kh, _Float16* __restrict__ Vh, int N4)
{
  int i = blockIdx.x * blockDim.x + threadIdx.x;
  if (i >= N4) return;
  float4 q = ((const float4*)Q)[i];
  float4 k = ((const float4*)K)[i];
  float4 v = ((const float4*)V)[i];
  h4 hq = {(_Float16)q.x, (_Float16)q.y, (_Float16)q.z, (_Float16)q.w};
  h4 hk = {(_Float16)k.x, (_Float16)k.y, (_Float16)k.z, (_Float16)k.w};
  h4 hv = {(_Float16)v.x, (_Float16)v.y, (_Float16)v.z, (_Float16)v.w};
  ((h4*)Qh)[i] = hq;
  ((h4*)Kh)[i] = hk;
  ((h4*)Vh)[i] = hv;
}

template <bool USE_TDM>
__global__ __launch_bounds__(256) void se_attn_kernel(
    const float* __restrict__ Qg_, const float* __restrict__ Kg_,
    const float* __restrict__ Vg_,
    const _Float16* __restrict__ Qh_, const _Float16* __restrict__ Kh_,
    const _Float16* __restrict__ Vh_,
    float* __restrict__ Og_, int S)
{
  constexpr int D = 64;
  // log2-domain score scale: log2(e) / (2*sqrt(d))
  constexpr float C2 = 1.44269504088896340736f / 16.0f;

  __shared__ alignas(16) _Float16 Qs[128 * QP];   // 128x64 Q tile (f16)
  __shared__ alignas(16) _Float16 Ks[32 * KP];    // 32x64 K tile (f16, row-major, padded)
  __shared__ alignas(16) _Float16 Vt[64 * VP];    // V tile transposed: Vt[d][key]
  __shared__ alignas(16) _Float16 Ps[8 * 16 * PP];// per-wave P bounce buffers
  __shared__ float k2s[32];                       // per-key ||k||^2 for current tile

  const int tid  = threadIdx.x;
  const int wave = tid >> 5;
  const int lane = tid & 31;
  const int hi   = lane >> 4;
  const int ln   = lane & 15;

  const int bh    = blockIdx.y;
  const int qbase = blockIdx.x * 128;

  const float*    Qg = Qg_ + (size_t)bh * S * D;
  const float*    Kg = Kg_ + (size_t)bh * S * D;
  const float*    Vg = Vg_ + (size_t)bh * S * D;
  const _Float16* Kh = USE_TDM ? Kh_ + (size_t)bh * S * D : nullptr;
  const _Float16* Vh = USE_TDM ? Vh_ + (size_t)bh * S * D : nullptr;
  float*          Og = Og_ + (size_t)bh * S * D;

  // ---- stage Q tile (128 x 64) ----
  if (USE_TDM) {
    if (wave == 0) {
      tdm_load_2d((uint32_t)(uintptr_t)&Qs[0],
                  Qh_ + (size_t)bh * S * D + (size_t)qbase * D,
                  /*tdim0*/64, /*tdim1*/128, /*tile0*/64, /*tile1*/128,
                  /*stride0*/64, /*pad: every 32 dw*/4, /*pad 4 dw*/3);
      __builtin_amdgcn_s_wait_tensorcnt(0);
    }
  } else {
#pragma unroll
    for (int pass = 0; pass < 8; ++pass) {
      int idx = pass * 256 + tid;
      int row = idx >> 4;
      int f4  = (idx & 15) * 4;
      float4 v = *(const float4*)&Qg[(size_t)(qbase + row) * D + f4];
      h4 hv = { (_Float16)v.x, (_Float16)v.y, (_Float16)v.z, (_Float16)v.w };
      *(h4*)&Qs[row * QP + f4] = hv;
    }
  }
  __syncthreads();

  // ---- ||q||^2 per row from the staged f16 values (one-time) ----
  float q2full = 0.f;
  {
    const _Float16* qr = &Qs[(wave * 16 + ln) * QP];
#pragma unroll
    for (int c = 0; c < 8; ++c) {
      h8 v = *(const h8*)(qr + c * 8);
#pragma unroll
      for (int j = 0; j < 8; ++j) { float f = (float)v[j]; q2full += f * f; }
    }
  }
  float q2v[8];
#pragma unroll
  for (int r = 0; r < 8; ++r) q2v[r] = __shfl(q2full, r + 8 * hi, 32);

  // ---- Q A-fragments (16x32 f16 each, d split into two K-chunks) ----
  const _Float16* qrow = &Qs[(wave * 16 + ln) * QP];
  h16 qa0 = ldfrag(qrow, hi * 8,      16 + hi * 8);
  h16 qa1 = ldfrag(qrow, 32 + hi * 8, 48 + hi * 8);

  float m[8], l[8];
#pragma unroll
  for (int r = 0; r < 8; ++r) { m[r] = -INFINITY; l[r] = 0.f; }
  f8 oAcc[4];
#pragma unroll
  for (int ct = 0; ct < 4; ++ct) oAcc[ct] = zero8();

  const int nt = S / 32;
  for (int kt = 0; kt < nt; ++kt) {
    const int kb0 = kt * 32;
    __syncthreads();

    if (USE_TDM) {
      // K tile via Tensor Data Mover (row padding in hardware)
      if (wave == 0)
        tdm_load_2d((uint32_t)(uintptr_t)&Ks[0], Kh + (size_t)kb0 * D,
                    64, 32, 64, 32, 64, 4, 3);
      // V tile: pure f16 copy + transpose
#pragma unroll
      for (int pass = 0; pass < 2; ++pass) {
        int idx = pass * 256 + tid;
        int key = idx >> 4;
        int f4  = (idx & 15) * 4;
        h4 hv = *(const h4*)&Vh[(size_t)(kb0 + key) * D + f4];
        Vt[(f4 + 0) * VP + key] = hv[0];
        Vt[(f4 + 1) * VP + key] = hv[1];
        Vt[(f4 + 2) * VP + key] = hv[2];
        Vt[(f4 + 3) * VP + key] = hv[3];
      }
      // cooperative ||k||^2: 16 lanes per key, shuffle-reduce, one store
#pragma unroll
      for (int pass = 0; pass < 2; ++pass) {
        int idx = pass * 256 + tid;
        int key = idx >> 4;
        int f4  = (idx & 15) * 4;
        h4 hv = *(const h4*)&Kh[(size_t)(kb0 + key) * D + f4];
        float p = (float)hv[0] * (float)hv[0] + (float)hv[1] * (float)hv[1]
                + (float)hv[2] * (float)hv[2] + (float)hv[3] * (float)hv[3];
        p += __shfl_xor(p, 1, 32);
        p += __shfl_xor(p, 2, 32);
        p += __shfl_xor(p, 4, 32);
        p += __shfl_xor(p, 8, 32);
        if ((lane & 15) == 0) k2s[key] = p;
      }
      if (kt + 1 < nt) {
        int nb = kb0 + 32;
        if (tid < 32)      __builtin_prefetch(&Kh[(size_t)(nb + tid) * D], 0, 0);
        else if (tid < 64) __builtin_prefetch(&Vh[(size_t)(nb + tid - 32) * D], 0, 0);
      }
      if (wave == 0) __builtin_amdgcn_s_wait_tensorcnt(0);
    } else {
      // fp32 staging with fused ||k||^2 reduction
#pragma unroll
      for (int pass = 0; pass < 2; ++pass) {
        int idx = pass * 256 + tid;
        int key = idx >> 4;
        int f4  = (idx & 15) * 4;
        float4 v = *(const float4*)&Kg[(size_t)(kb0 + key) * D + f4];
        h4 hv = { (_Float16)v.x, (_Float16)v.y, (_Float16)v.z, (_Float16)v.w };
        *(h4*)&Ks[key * KP + f4] = hv;
        float p = v.x * v.x + v.y * v.y + v.z * v.z + v.w * v.w;
        p += __shfl_xor(p, 1, 32);
        p += __shfl_xor(p, 2, 32);
        p += __shfl_xor(p, 4, 32);
        p += __shfl_xor(p, 8, 32);
        if ((lane & 15) == 0) k2s[key] = p;
      }
#pragma unroll
      for (int pass = 0; pass < 2; ++pass) {
        int idx = pass * 256 + tid;
        int key = idx >> 4;
        int f4  = (idx & 15) * 4;
        float4 v = *(const float4*)&Vg[(size_t)(kb0 + key) * D + f4];
        Vt[(f4 + 0) * VP + key] = (_Float16)v.x;
        Vt[(f4 + 1) * VP + key] = (_Float16)v.y;
        Vt[(f4 + 2) * VP + key] = (_Float16)v.z;
        Vt[(f4 + 3) * VP + key] = (_Float16)v.w;
      }
      if (kt + 1 < nt) {
        int nb = kb0 + 32;
        if (tid < 32)      __builtin_prefetch(&Kg[(size_t)(nb + tid) * D], 0, 0);
        else if (tid < 64) __builtin_prefetch(&Vg[(size_t)(nb + tid - 32) * D], 0, 0);
      }
    }
    __syncthreads();

    const float k2t0 = k2s[ln];
    const float k2t1 = k2s[16 + ln];

    // ---- S = Q K^T : two 16x16 tiles, 2 WMMAs each ----
    f8 sacc[2];
#pragma unroll
    for (int t = 0; t < 2; ++t) {
      const _Float16* krow = &Ks[(t * 16 + ln) * KP];
      h16 kbf0 = ldfrag(krow, hi * 8,      16 + hi * 8);
      h16 kbf1 = ldfrag(krow, 32 + hi * 8, 48 + hi * 8);
      f8 c = zero8();
      c = __builtin_amdgcn_wmma_f32_16x16x32_f16(false, qa0, false, kbf0, (short)0, c, false, false);
      c = __builtin_amdgcn_wmma_f32_16x16x32_f16(false, qa1, false, kbf1, (short)0, c, false, false);
      sacc[t] = c;
    }

    // ---- SE logit transform, directly in log2 domain (raw v_sqrt_f32) ----
    float sc[2][8];
#pragma unroll
    for (int t = 0; t < 2; ++t) {
      const float k2 = (t == 0) ? k2t0 : k2t1;
#pragma unroll
      for (int r = 0; r < 8; ++r) {
        float bsum = q2v[r] + k2;
        float sq   = fmaxf(bsum - 2.f * sacc[t][r], 0.f);   // ||q-k||^2 (clamped)
        sc[t][r]   = (bsum - fast_sqrt(sq)) * C2;           // log2-scaled logit
      }
    }

    // ---- online softmax: reductions across 16-lane half-groups ----
    float mt[8];
#pragma unroll
    for (int r = 0; r < 8; ++r) mt[r] = fmaxf(sc[0][r], sc[1][r]);
#pragma unroll
    for (int mask = 1; mask <= 8; mask <<= 1)
#pragma unroll
      for (int r = 0; r < 8; ++r) mt[r] = fmaxf(mt[r], __shfl_xor(mt[r], mask, 32));

    float alpha[8], rs[8];
    _Float16* pbase = &Ps[wave * 16 * PP];
#pragma unroll
    for (int r = 0; r < 8; ++r) {
      float mn = fmaxf(m[r], mt[r]);
      alpha[r] = fast_exp2(m[r] - mn);
      m[r] = mn;
      rs[r] = 0.f;
#pragma unroll
      for (int t = 0; t < 2; ++t) {
        float p = fast_exp2(sc[t][r] - mn);
        rs[r] += p;
        pbase[(r + 8 * hi) * PP + t * 16 + ln] = (_Float16)p;  // C-layout -> LDS
      }
    }
#pragma unroll
    for (int mask = 1; mask <= 8; mask <<= 1)
#pragma unroll
      for (int r = 0; r < 8; ++r) rs[r] += __shfl_xor(rs[r], mask, 32);
#pragma unroll
    for (int r = 0; r < 8; ++r) l[r] = l[r] * alpha[r] + rs[r];

#pragma unroll
    for (int ct = 0; ct < 4; ++ct)
#pragma unroll
      for (int r = 0; r < 8; ++r) oAcc[ct][r] *= alpha[r];

    // wave-private LDS bounce: wait for this wave's P stores
    asm volatile("s_wait_dscnt 0" ::: "memory");

    // ---- O += P V ----
    const _Float16* prow = &Ps[(wave * 16 + ln) * PP];
    h16 pa = ldfrag(prow, hi * 8, 16 + hi * 8);
#pragma unroll
    for (int ct = 0; ct < 4; ++ct) {
      const _Float16* vcol = &Vt[(ct * 16 + ln) * VP];
      h16 vb = ldfrag(vcol, hi * 8, 16 + hi * 8);
      oAcc[ct] = __builtin_amdgcn_wmma_f32_16x16x32_f16(false, pa, false, vb, (short)0, oAcc[ct], false, false);
    }
  }

  // ---- finalize ----
#pragma unroll
  for (int r = 0; r < 8; ++r) {
    float inv = fast_rcp(l[r]);
    int rowg = qbase + wave * 16 + r + 8 * hi;
    float* orow = Og + (size_t)rowg * D;
#pragma unroll
    for (int ct = 0; ct < 4; ++ct)
      orow[ct * 16 + ln] = oAcc[ct][r] * inv;
  }
}

extern "C" void kernel_launch(void* const* d_in, const int* in_sizes, int n_in,
                              void* d_out, int out_size, void* d_ws, size_t ws_size,
                              hipStream_t stream) {
  (void)n_in; (void)out_size;
  constexpr int S = 2048, D = 64;
  const int BH = in_sizes[0] / (S * D);       // B*H = 32
  const size_t halves = (size_t)BH * S * D;   // elements per tensor
  dim3 grid(S / 128, BH, 1);

  const float* Q = (const float*)d_in[0];
  const float* K = (const float*)d_in[1];
  const float* V = (const float*)d_in[2];
  float*       O = (float*)d_out;

  if (ws_size >= 3 * halves * sizeof(_Float16)) {
    _Float16* Qh = (_Float16*)d_ws;
    _Float16* Kh = Qh + halves;
    _Float16* Vh = Kh + halves;
    const int N4 = (int)(halves / 4);
    cvt3_kernel<<<dim3((N4 + 255) / 256), 256, 0, stream>>>(Q, K, V, Qh, Kh, Vh, N4);
    se_attn_kernel<true><<<grid, 256, 0, stream>>>(Q, K, V, Qh, Kh, Vh, O, S);
  } else {
    se_attn_kernel<false><<<grid, 256, 0, stream>>>(Q, K, V, nullptr, nullptr, nullptr, O, S);
  }
}
